// MiniMoE_69973607186434
// MI455X (gfx1250) — matrix operand order repaired and checked by reading, not compiled, not run on hardware
//
#include <hip/hip_runtime.h>

// ---------------- problem constants (B=4, S=2048, D=1024, E=8, dff=4096) ----
#define T_TOK 8192
#define DIM   1024
#define DFF   4096
#define NEXP  8

// ---------------- vector types ---------------------------------------------
typedef __attribute__((ext_vector_type(16))) __bf16          v16bf;
typedef __attribute__((ext_vector_type(8)))  float           v8f;
typedef __attribute__((ext_vector_type(8)))  unsigned short  u16x8;
typedef __attribute__((ext_vector_type(16))) unsigned short  u16x16;
typedef __attribute__((ext_vector_type(16))) float           f32x16;
typedef __attribute__((ext_vector_type(8)))  float           f32x8;

// ---------------- fp32 -> bf16 split helpers --------------------------------
__device__ __forceinline__ unsigned short bf16_rne(float f) {
  unsigned int u = __builtin_bit_cast(unsigned int, f);
  u += 0x7FFFu + ((u >> 16) & 1u);
  return (unsigned short)(u >> 16);
}
__device__ __forceinline__ float bf16_to_f32(unsigned short h) {
  return __builtin_bit_cast(float, (unsigned int)h << 16);
}

// A-fragment: lane(m, khalf) holds 16 contiguous K values starting at
// khalf*16 (matches fragB_ld).
__device__ __forceinline__ v16bf fragA_ld(const unsigned short* row, int kh) {
  union { struct { u16x8 a, b; } s; v16bf v; } u;
  u.s.a = *(const u16x8*)(row + kh * 16);
  u.s.b = *(const u16x8*)(row + kh * 16 + 8);
  return u.v;
}
// B-fragment, same contiguous K order as fragA_ld: lane(n, khalf) holds 16
// contiguous K values starting at khalf*16.
__device__ __forceinline__ v16bf fragB_ld(const unsigned short* row, int kh) {
  union { u16x16 raw; v16bf v; } u;
  u.raw = *(const u16x16*)(row + kh * 16);
  return u.v;
}

__device__ __forceinline__ v8f wmma_bf16(v16bf a, v16bf b, v8f c) {
  return __builtin_amdgcn_wmma_f32_16x16x32_bf16(
      /*neg_a=*/false, a, /*neg_b=*/false, b,
      /*c_mod=*/(short)0, c, /*reuse_a=*/false, /*reuse_b=*/false);
}

// ============================================================================
// GEMM: C[M,N] (+)= act( A[M,K] @ B[N,K]^T ), fp32 in/out, bf16x2-split WMMA.
// Workgroup = 256 threads = 8 waves; tile 128(M) x 64(N); K-step 32.
// actMode: 0 = store, 1 = relu^2 store, 2 = accumulate (+=).
// ============================================================================
__global__ __launch_bounds__(256) void gemm_bf16x2(
    const float* __restrict__ A, const float* __restrict__ B,
    float* __restrict__ C,
    const int* __restrict__ gatherIds, const int* __restrict__ gatherOfsPtr,
    const int* __restrict__ McntPtr, int Mconst,
    const int* __restrict__ outOfsPtr,
    int N, int K, int actMode)
{
  const int M = McntPtr ? McntPtr[0] : Mconst;
  const int base = blockIdx.y * 128;
  if (base >= M) return;                       // ragged-M early exit (uniform)
  const int nbase  = blockIdx.x * 64;
  const int gofs   = gatherOfsPtr ? gatherOfsPtr[0] : 0;
  const int outOfs = outOfsPtr ? outOfsPtr[0] : 0;
  const int tid = threadIdx.x;

  __shared__ __align__(64) unsigned short sAhi[128][32];
  __shared__ __align__(64) unsigned short sAlo[128][32];
  __shared__ __align__(64) unsigned short sBhi[64][32];
  __shared__ __align__(64) unsigned short sBlo[64][32];

  // --- staging assignment: 2 threads per A row (16 floats each),
  //     4 threads per B row (8 floats each) -----------------------------
  const int rA = tid >> 1, hA = tid & 1;
  int rr = base + rA; if (rr > M - 1) rr = M - 1;         // clamp ragged rows
  const int rowA = gatherIds ? gatherIds[gofs + rr] : rr; // token gather
  const float* aPtr = A + (size_t)rowA * K + hA * 16;
  const int rB = tid >> 2, qB = tid & 3;
  const float* bPtr = B + (size_t)(nbase + rB) * K + qB * 8;

  const int w = tid >> 5, L = tid & 31, kh = L >> 4, mL = L & 15;
  const unsigned short* aHiRow = &sAhi[w * 16 + mL][0];
  const unsigned short* aLoRow = &sAlo[w * 16 + mL][0];

  v8f acc[4];
  {
    v8f z = {0.f,0.f,0.f,0.f,0.f,0.f,0.f,0.f};
    #pragma unroll
    for (int i = 0; i < 4; ++i) acc[i] = z;
  }

  for (int kb = 0; kb < K; kb += 32) {
    f32x16 av = *(const f32x16*)(aPtr + kb);
    f32x8  bv = *(const f32x8*)(bPtr + kb);
    if (kb + 32 < K) {                      // prefetch next K tile (L2 hot)
      __builtin_prefetch(aPtr + kb + 32, 0, 1);
      __builtin_prefetch(bPtr + kb + 32, 0, 1);
    }
    // split fp32 -> (bf16 hi, bf16 lo) : x ~= hi + lo, |lo| <= 2^-8 |x|
    u16x8 ah[2], al[2];
    #pragma unroll
    for (int j = 0; j < 16; ++j) {
      float x = av[j];
      unsigned short h = bf16_rne(x);
      unsigned short l = bf16_rne(x - bf16_to_f32(h));
      ah[j >> 3][j & 7] = h; al[j >> 3][j & 7] = l;
    }
    u16x8 bh, bl;
    #pragma unroll
    for (int j = 0; j < 8; ++j) {
      float x = bv[j];
      unsigned short h = bf16_rne(x);
      unsigned short l = bf16_rne(x - bf16_to_f32(h));
      bh[j] = h; bl[j] = l;
    }
    *(u16x8*)&sAhi[rA][hA * 16 + 0] = ah[0];
    *(u16x8*)&sAhi[rA][hA * 16 + 8] = ah[1];
    *(u16x8*)&sAlo[rA][hA * 16 + 0] = al[0];
    *(u16x8*)&sAlo[rA][hA * 16 + 8] = al[1];
    *(u16x8*)&sBhi[rB][qB * 8] = bh;
    *(u16x8*)&sBlo[rB][qB * 8] = bl;
    __syncthreads();

    v16bf aHi = fragA_ld(aHiRow, kh);
    v16bf aLo = fragA_ld(aLoRow, kh);
    #pragma unroll
    for (int nb = 0; nb < 4; ++nb) {
      v16bf bHi = fragB_ld(&sBhi[nb * 16 + mL][0], kh);
      v16bf bLo = fragB_ld(&sBlo[nb * 16 + mL][0], kh);
      // 2-term split product: hi*hi + hi*lo + lo*hi (lo*lo ~ 2^-18, dropped)
      acc[nb] = wmma_bf16(aLo, bHi, acc[nb]);
      acc[nb] = wmma_bf16(aHi, bLo, acc[nb]);
      acc[nb] = wmma_bf16(aHi, bHi, acc[nb]);
    }
    __syncthreads();
  }

  // epilogue per CDNA5 C layout: lanes 0-15 -> M=j, lanes 16-31 -> M=j+8
  #pragma unroll
  for (int nb = 0; nb < 4; ++nb) {
    const int col = nbase + nb * 16 + mL;
    #pragma unroll
    for (int j = 0; j < 8; ++j) {
      const int m = j + 8 * kh;
      const int r = base + w * 16 + m;
      if (r < M) {
        float v = acc[nb][j];
        if (actMode == 1) { v = fmaxf(v, 0.0f); v = v * v; }
        float* dst = C + (size_t)(outOfs + r) * N + col;
        if (actMode == 2) *dst += v; else *dst = v;
      }
    }
  }
}

// ============================================================================
// Router: one wave per token. logits = x @ rw^T, softmax, top-2, renorm.
// ============================================================================
__global__ __launch_bounds__(256) void router_topk(
    const float* __restrict__ x, const float* __restrict__ rw,
    int* __restrict__ counts, int* __restrict__ tokExp, float* __restrict__ tokW)
{
  const int t = blockIdx.x * 8 + (threadIdx.x >> 5);
  const int lane = threadIdx.x & 31;
  const float* xr = x + (size_t)t * DIM;
  float acc[NEXP];
  #pragma unroll
  for (int e = 0; e < NEXP; ++e) acc[e] = 0.0f;
  for (int j = 0; j < DIM / 32; ++j) {
    const int k = lane + 32 * j;
    const float xv = xr[k];
    #pragma unroll
    for (int e = 0; e < NEXP; ++e)
      acc[e] = __builtin_fmaf(xv, rw[e * DIM + k], acc[e]);
  }
  #pragma unroll
  for (int e = 0; e < NEXP; ++e) {
    #pragma unroll
    for (int off = 16; off > 0; off >>= 1)
      acc[e] += __shfl_xor(acc[e], off, 32);
  }
  if (lane == 0) {
    float mx = acc[0];
    #pragma unroll
    for (int e = 1; e < NEXP; ++e) mx = fmaxf(mx, acc[e]);
    float p[NEXP];
    #pragma unroll
    for (int e = 0; e < NEXP; ++e) p[e] = __expf(acc[e] - mx);
    // top-2 (first-occurrence tie-break, matches lax.top_k); softmax
    // denominator cancels in the top-2 renormalization.
    int i0 = 0;
    #pragma unroll
    for (int e = 1; e < NEXP; ++e) if (p[e] > p[i0]) i0 = e;
    int i1 = (i0 == 0) ? 1 : 0;
    #pragma unroll
    for (int e = 0; e < NEXP; ++e) if (e != i0 && p[e] > p[i1]) i1 = e;
    const float s = fmaxf(p[i0] + p[i1], 1e-30f);
    atomicAdd(&counts[i0], 1);
    atomicAdd(&counts[i1], 1);
    tokExp[2 * t + 0] = i0;  tokW[2 * t + 0] = p[i0] / s;
    tokExp[2 * t + 1] = i1;  tokW[2 * t + 1] = p[i1] / s;
  }
}

__global__ void scan_offsets(const int* __restrict__ counts,
                             int* __restrict__ offsets, int* __restrict__ cursor)
{
  if (threadIdx.x == 0 && blockIdx.x == 0) {
    int s = 0;
    for (int e = 0; e < NEXP; ++e) { offsets[e] = s; cursor[e] = s; s += counts[e]; }
    offsets[NEXP] = s;
  }
}

__global__ __launch_bounds__(256) void assign_slots(
    const int* __restrict__ tokExp, int* __restrict__ cursor,
    int* __restrict__ slots, int* __restrict__ tokenIds)
{
  const int t = blockIdx.x * 256 + threadIdx.x;
  #pragma unroll
  for (int k = 0; k < 2; ++k) {
    const int e = tokExp[2 * t + k];
    const int pos = atomicAdd(&cursor[e], 1);
    slots[2 * t + k] = pos;
    tokenIds[pos] = t;
  }
}

// out[t] = w0*ybuf[slot0] + w1*ybuf[slot1]   (shared expert added afterwards)
__global__ __launch_bounds__(256) void combine_topk(
    const float* __restrict__ ybuf, const int* __restrict__ slots,
    const float* __restrict__ tokW, float* __restrict__ out)
{
  const int idx = blockIdx.x * 256 + threadIdx.x;   // over T*D/4 float4s
  const int t = idx >> 8;                            // D/4 = 256
  const int d4 = idx & 255;
  const float4* yb = (const float4*)ybuf;
  const float4 a = yb[(size_t)slots[2 * t + 0] * 256 + d4];
  const float4 b = yb[(size_t)slots[2 * t + 1] * 256 + d4];
  const float w0 = tokW[2 * t + 0], w1 = tokW[2 * t + 1];
  float4 o;
  o.x = w0 * a.x + w1 * b.x;
  o.y = w0 * a.y + w1 * b.y;
  o.z = w0 * a.z + w1 * b.z;
  o.w = w0 * a.w + w1 * b.w;
  ((float4*)out)[idx] = o;
}

// ============================================================================
extern "C" void kernel_launch(void* const* d_in, const int* in_sizes, int n_in,
                              void* d_out, int out_size, void* d_ws, size_t ws_size,
                              hipStream_t stream) {
  (void)in_sizes; (void)n_in; (void)out_size; (void)ws_size;
  const float* x     = (const float*)d_in[0];   // [T, D]
  const float* rw    = (const float*)d_in[1];   // [E, D]
  const float* wfc   = (const float*)d_in[2];   // [E, DFF, D]
  const float* wproj = (const float*)d_in[3];   // [E, D, DFF]
  const float* sfc   = (const float*)d_in[4];   // [DFF, D]
  const float* sproj = (const float*)d_in[5];   // [D, DFF]
  float* out = (float*)d_out;                   // [T, D]

  // ---- workspace carving (256B aligned) ----
  char* p = (char*)d_ws; size_t o = 0;
  auto take = [&](size_t bytes) -> void* {
    void* r = p + o; o += (bytes + 255) & ~(size_t)255; return r;
  };
  int*   counts   = (int*)take(NEXP * sizeof(int));
  int*   offsets  = (int*)take((NEXP + 1) * sizeof(int));
  int*   cursor   = (int*)take(NEXP * sizeof(int));
  int*   tokExp   = (int*)take((size_t)2 * T_TOK * sizeof(int));
  float* tokW     = (float*)take((size_t)2 * T_TOK * sizeof(float));
  int*   slots    = (int*)take((size_t)2 * T_TOK * sizeof(int));
  int*   tokenIds = (int*)take((size_t)2 * T_TOK * sizeof(int));
  float* h        = (float*)take((size_t)T_TOK * DFF * sizeof(float));   // 128 MB
  float* ybuf     = (float*)take((size_t)2 * T_TOK * DIM * sizeof(float)); // 64 MB

  hipMemsetAsync(counts, 0, NEXP * sizeof(int), stream);
  router_topk<<<T_TOK / 8, 256, 0, stream>>>(x, rw, counts, tokExp, tokW);
  scan_offsets<<<1, 32, 0, stream>>>(counts, offsets, cursor);
  assign_slots<<<T_TOK / 256, 256, 0, stream>>>(tokExp, cursor, slots, tokenIds);

  // Per-expert sparse MLP: expert-sequential so each expert's 33MB weight set
  // stays hot in the 192MB L2 across all of its token tiles.
  const dim3 gridFC(DFF / 64, T_TOK / 128);
  const dim3 gridPJ(DIM / 64, T_TOK / 128);
  for (int e = 0; e < NEXP; ++e) {
    gemm_bf16x2<<<gridFC, 256, 0, stream>>>(
        x, wfc + (size_t)e * DFF * DIM, h,
        tokenIds, &offsets[e], &counts[e], 0, nullptr,
        DFF, DIM, /*act=*/1);
    gemm_bf16x2<<<gridPJ, 256, 0, stream>>>(
        h, wproj + (size_t)e * DIM * DFF, ybuf,
        nullptr, nullptr, &counts[e], 0, &offsets[e],
        DIM, DFF, /*act=*/0);
  }

  combine_topk<<<(T_TOK * DIM / 4) / 256, 256, 0, stream>>>(ybuf, slots, tokW, out);

  // Shared expert on all tokens, accumulated into the combined output.
  gemm_bf16x2<<<gridFC, 256, 0, stream>>>(
      x, sfc, h, nullptr, nullptr, nullptr, T_TOK, nullptr, DFF, DIM, /*act=*/1);
  gemm_bf16x2<<<gridPJ, 256, 0, stream>>>(
      h, sproj, out, nullptr, nullptr, nullptr, T_TOK, nullptr, DIM, DFF, /*act=*/2);
}
